// DualBrainBlock_51539608082
// MI455X (gfx1250) — matrix operand order repaired and checked by reading, not compiled, not run on hardware
//
#include <hip/hip_runtime.h>
#include <hip/hip_bf16.h>

typedef __attribute__((ext_vector_type(16))) _Float16 v16h;
typedef __attribute__((ext_vector_type(8)))  float    v8f;

#define HD   1024
#define SEQ  4096
#define BATCH 16
#define NEXP 4
#define CHUNK 64            // timesteps per chunk
#define NCHUNK (SEQ / CHUNK)
#define EPS 1e-6f

// ---------------------------------------------------------------- kernel 0
__global__ void k_zero(float* acc) {
    int i = blockIdx.x * blockDim.x + threadIdx.x;
    if (i < BATCH * HD) acc[i] = 0.0f;
}

// ---------------------------------------------------------------- kernel 1
// Fused: per-row rmsnorm scale + weighted EMA reduction over one chunk.
// state_{S-1}[b,d] = sum_t (1-dec[d]) * dec[d]^(S-1-t) * xnorm[b,t,d]
// grid = (NCHUNK, BATCH), block = 256 (8 waves). Each thread owns 4 channels.
__global__ __launch_bounds__(256) void k_scan(const float* __restrict__ x,
                                              const float* __restrict__ decay_logit,
                                              float* __restrict__ acc) {
    __shared__ float rms_lds[CHUNK];
    const int chunk = blockIdx.x;
    const int b     = blockIdx.y;
    const int tid   = threadIdx.x;
    const int lane  = tid & 31;
    const int wave  = tid >> 5;
    const int t0    = chunk * CHUNK;
    const int t1    = t0 + CHUNK;
    const size_t xb = (size_t)b * SEQ * HD;

    // ---- phase 1: rms_inv for the 64 rows of this chunk (8 rows per wave)
    // vectorized: lane reads float4 at k*512B + lane*16B -> global_load_b128,
    // each step the wave covers 512 contiguous bytes.
    for (int j = 0; j < CHUNK / 8; ++j) {
        int r = wave + 8 * j;                 // local row
        const float4* row4 = (const float4*)(x + xb + (size_t)(t0 + r) * HD);
        float s = 0.0f;
        #pragma unroll
        for (int k = 0; k < HD / 128; ++k) {
            float4 v = row4[k * 32 + lane];
            s += v.x * v.x + v.y * v.y + v.z * v.z + v.w * v.w;
        }
        #pragma unroll
        for (int off = 16; off > 0; off >>= 1)
            s += __shfl_xor(s, off, 32);
        if (lane == 0) rms_lds[r] = rsqrtf(s * (1.0f / HD) + EPS);
    }
    __syncthreads();

    // ---- phase 2: weighted accumulation, 4 channels per thread.
    // Iterate t downward so the running weight decays (multiply by dec).
    const int d0 = tid * 4;
    float dec[4], w[4], a[4];
    #pragma unroll
    for (int j = 0; j < 4; ++j) {
        float z = decay_logit[d0 + j];
        dec[j] = 1.0f / (1.0f + __expf(-z));           // sigmoid
        w[j]   = __powf(dec[j], (float)(SEQ - t1));    // dec^(S-1-t) at t=t1-1
        a[j]   = 0.0f;
    }
    const float4* xrow4 = (const float4*)(x + xb);
    for (int t = t1 - 1; t >= t0; --t) {
        if (t - 2 >= t0)                      // descending walk: prefetch ahead
            __builtin_prefetch(&xrow4[(size_t)(t - 2) * (HD / 4) + tid], 0, 1);
        float rinv = rms_lds[t - t0];
        float4 xv = xrow4[(size_t)t * (HD / 4) + tid];
        a[0] += xv.x * rinv * w[0];
        a[1] += xv.y * rinv * w[1];
        a[2] += xv.z * rinv * w[2];
        a[3] += xv.w * rinv * w[3];
        #pragma unroll
        for (int j = 0; j < 4; ++j) w[j] *= dec[j];
    }
    #pragma unroll
    for (int j = 0; j < 4; ++j)
        atomicAdd(&acc[b * HD + d0 + j], a[j]);
}

// ---------------------------------------------------------------- kernel 2
// final = x[:, S-1, :] + acc*(1-decay)*norm1_w ; pool = rmsnorm(final, norm2_w) -> f16
// grid = BATCH, block = 256, 4 channels per thread.
__global__ __launch_bounds__(256) void k_pool(const float* __restrict__ x,
                                              const float* __restrict__ acc,
                                              const float* __restrict__ decay_logit,
                                              const float* __restrict__ n1w,
                                              const float* __restrict__ n2w,
                                              _Float16* __restrict__ pool_h) {
    __shared__ float red[8];
    __shared__ float rinv_sh;
    const int b   = blockIdx.x;
    const int tid = threadIdx.x;
    const int d0  = tid * 4;
    const float4* xlast4 = (const float4*)(x + ((size_t)b * SEQ + (SEQ - 1)) * HD);

    float4 xv = xlast4[tid];
    float xl[4] = {xv.x, xv.y, xv.z, xv.w};
    float fin[4];
    float ss = 0.0f;
    #pragma unroll
    for (int j = 0; j < 4; ++j) {
        int d = d0 + j;
        float dc = 1.0f / (1.0f + __expf(-decay_logit[d]));
        float st = acc[b * HD + d] * (1.0f - dc) * n1w[d];
        fin[j] = xl[j] + st;
        ss += fin[j] * fin[j];
    }
    #pragma unroll
    for (int off = 16; off > 0; off >>= 1)
        ss += __shfl_xor(ss, off, 32);
    if ((tid & 31) == 0) red[tid >> 5] = ss;
    __syncthreads();
    if (tid == 0) {
        float tot = 0.0f;
        #pragma unroll
        for (int wv = 0; wv < 8; ++wv) tot += red[wv];
        rinv_sh = rsqrtf(tot * (1.0f / HD) + EPS);
    }
    __syncthreads();
    float rinv = rinv_sh;
    #pragma unroll
    for (int j = 0; j < 4; ++j) {
        int d = d0 + j;
        pool_h[b * HD + d] = (_Float16)(fin[j] * rinv * n2w[d]);
    }
}

// ---------------------------------------------------------------- kernel 3
// out[b,o] = relu( pool[b,:] . W[experts[b], o, :] ) via WMMA, computed for
// every expert; epilogue keeps only rows whose expert matches this tile.
// grid = (HD/16 o-tiles, NEXP), block = 32 (one wave per 16x16 tile).
__global__ __launch_bounds__(32) void k_gemm(const _Float16* __restrict__ pool_h,
                                             const float* __restrict__ W,
                                             const int* __restrict__ experts,
                                             float* __restrict__ out) {
    const int lane = threadIdx.x;      // 0..31
    const int n    = lane & 15;        // column within tile (N) / row M for A
    const int hi   = lane >> 4;        // half-wave selector
    const int o0   = blockIdx.x * 16;
    const int e    = blockIdx.y;
    const float* Wrow = W + ((size_t)e * HD + (o0 + n)) * HD;  // row-major in K

    v8f c = {};
    for (int kb = 0; kb < HD; kb += 32) {
        v16h a, bm;
        // A (pool) 16x32 f16 layout: lanes<16 -> K {0..7,16..23}; lanes>=16 -> +8
        #pragma unroll
        for (int i = 0; i < 16; ++i) {
            int koff = (i < 8 ? i : i + 8) + hi * 8;
            a[i] = pool_h[n * HD + kb + koff];
        }
        // B (W tile) 32x16 f16 layout: lanes<16 -> K {0..7,16..23}; lanes>=16 -> +8
        #pragma unroll
        for (int i = 0; i < 16; ++i) {
            bm[i] = (_Float16)Wrow[kb + (i < 8 ? i : i + 8) + hi * 8];
        }
        c = __builtin_amdgcn_wmma_f32_16x16x32_f16(
                /*neg_a=*/false, a, /*neg_b=*/false, bm,
                /*c_mod=*/(short)0, c, /*reuse_a=*/false, /*reuse_b=*/false);
    }
    // C/D layout: VGPR r holds M = r (lanes 0-15) or M = r+8 (lanes 16-31)
    #pragma unroll
    for (int r = 0; r < 8; ++r) {
        int m = r + hi * 8;                   // batch row
        if (experts[m] == e) {
            float v = c[r];
            out[m * HD + o0 + n] = v > 0.0f ? v : 0.0f;
        }
    }
}

// ---------------------------------------------------------------- launch
extern "C" void kernel_launch(void* const* d_in, const int* in_sizes, int n_in,
                              void* d_out, int out_size, void* d_ws, size_t ws_size,
                              hipStream_t stream) {
    (void)in_sizes; (void)n_in; (void)out_size; (void)ws_size;
    const float* x      = (const float*)d_in[0];
    const int*   exps   = (const int*)d_in[1];
    const float* n1w    = (const float*)d_in[2];
    const float* dlogit = (const float*)d_in[3];
    const float* n2w    = (const float*)d_in[4];
    const float* W      = (const float*)d_in[5];
    float* out = (float*)d_out;

    float*    acc    = (float*)d_ws;                              // 16*1024 f32
    _Float16* pool_h = (_Float16*)((char*)d_ws + BATCH * HD * 4); // 16*1024 f16

    k_zero<<<dim3((BATCH * HD + 255) / 256), dim3(256), 0, stream>>>(acc);
    k_scan<<<dim3(NCHUNK, BATCH), dim3(256), 0, stream>>>(x, dlogit, acc);
    k_pool<<<dim3(BATCH), dim3(256), 0, stream>>>(x, acc, dlogit, n1w, n2w, pool_h);
    k_gemm<<<dim3(HD / 16, NEXP), dim3(32), 0, stream>>>(pool_h, W, exps, out);
}
